// PVConvReLU_23313082483563
// MI455X (gfx1250) — compile-verified, matrix operand order checked by reading, not run on hardware
//
#include <hip/hip_runtime.h>
#include <hip/hip_bf16.h>
#include <stdint.h>

#define RES    32
#define R3     32768
#define BATCH  8
#define NPTS   32768
#define CCH    64
#define BN_EPS 1e-5f
#define LEAK   0.2f

typedef __attribute__((ext_vector_type(16))) __bf16 bf16x16;
typedef __attribute__((ext_vector_type(8)))  float  floatx8;

union FragAB { uint32_t u[8]; uint4 q[2]; bf16x16 v; };
union Pack8  { uint4 q; __bf16 h[8]; };

// ---------------- voxelization: per-batch coord mean ----------------
__global__ void k_coord_mean(const float* __restrict__ coords, float* __restrict__ mean) {
    __shared__ float red[256];
    int b = blockIdx.x, tid = threadIdx.x;
    for (int ax = 0; ax < 3; ++ax) {
        float s = 0.f;
        const float* src = coords + ((size_t)b * 3 + ax) * NPTS;
        for (int n = tid; n < NPTS; n += 256) s += src[n];
        red[tid] = s; __syncthreads();
        for (int o = 128; o > 0; o >>= 1) { if (tid < o) red[tid] += red[tid + o]; __syncthreads(); }
        if (tid == 0) mean[b * 3 + ax] = red[0] * (1.f / NPTS);
        __syncthreads();
    }
}

// ---------------- per-batch max norm -> rscale = 16/M ----------------
__global__ void k_coord_scale(const float* __restrict__ coords, const float* __restrict__ mean,
                              float* __restrict__ rscale) {
    __shared__ float red[256];
    int b = blockIdx.x, tid = threadIdx.x;
    float m0 = mean[b * 3 + 0], m1 = mean[b * 3 + 1], m2 = mean[b * 3 + 2];
    const float* cx = coords + (size_t)b * 3 * NPTS;
    float mx = 0.f;
    for (int n = tid; n < NPTS; n += 256) {
        float dx = cx[n] - m0;
        float dy = cx[NPTS + n] - m1;
        float dz = cx[2 * NPTS + n] - m2;
        mx = fmaxf(mx, dx * dx + dy * dy + dz * dz);
    }
    red[tid] = mx; __syncthreads();
    for (int o = 128; o > 0; o >>= 1) { if (tid < o) red[tid] = fmaxf(red[tid], red[tid + o]); __syncthreads(); }
    if (tid == 0) rscale[b] = 16.f / sqrtf(red[0]);   // R/(2*maxnorm)
}

// ---------------- scatter-mean into voxel grid (sum + count) ----------------
__global__ void k_voxelize(const float* __restrict__ feat, const float* __restrict__ coords,
                           const float* __restrict__ mean, const float* __restrict__ rscale,
                           float* __restrict__ nc_out,          // [B*N][3]
                           float* __restrict__ gsum,            // [B][R3][64]
                           float* __restrict__ cnt) {           // [B][R3]
    int t = blockIdx.x * 256 + threadIdx.x;
    if (t >= BATCH * NPTS) return;
    int b = t / NPTS, n = t - b * NPTS;
    float rs = rscale[b];
    int vi[3];
    #pragma unroll
    for (int ax = 0; ax < 3; ++ax) {
        float x = (coords[((size_t)b * 3 + ax) * NPTS + n] - mean[b * 3 + ax]) * rs + 16.f;
        x = fminf(fmaxf(x, 0.f), 31.f);
        nc_out[(size_t)t * 3 + ax] = x;
        vi[ax] = (int)rintf(x);                       // round-half-even == jnp.round
    }
    int v = (vi[0] * RES + vi[1]) * RES + vi[2];
    atomicAdd(&cnt[(size_t)b * R3 + v], 1.f);
    float* gs = &gsum[((size_t)b * R3 + v) * CCH];
    for (int c = 0; c < CCH; ++c)
        atomicAdd(&gs[c], feat[((size_t)b * CCH + c) * NPTS + n]);
}

// ---------------- average and convert to bf16 grid [B][C][R3] ----------------
__global__ void k_finalize_grid(const float* __restrict__ gsum, const float* __restrict__ cnt,
                                __bf16* __restrict__ g0) {
    int t = blockIdx.x * 256 + threadIdx.x;           // B*R3
    if (t >= BATCH * R3) return;
    int b = t / R3, v = t - b * R3;
    float inv = 1.f / fmaxf(cnt[t], 1.f);
    const float* gs = &gsum[(size_t)t * CCH];
    for (int c = 0; c < CCH; ++c)
        g0[((size_t)(b * CCH + c)) * R3 + v] = (__bf16)(gs[c] * inv);
}

// ---------------- pack conv weights into WMMA A-fragment layout ----------------
// wpack[tap27][kc2][mtile4][lane32][v8] : bf16 pair per dword, per ISA 16-bit A layout
__global__ void k_pack_conv(const float* __restrict__ w, uint32_t* __restrict__ wpack) {
    int t = blockIdx.x * 256 + threadIdx.x;
    if (t >= 27 * 2 * 4 * 32 * 8) return;
    int v    = t & 7;
    int lane = (t >> 3) & 31;
    int mt   = (t >> 8) & 3;
    int kc   = (t >> 10) & 1;
    int tap  = t >> 11;
    int m = mt * 16 + (lane & 15);
    int kbase = ((v & 4) << 2) + ((lane >= 16) ? 8 : 0) + 2 * (v & 3);
    uint32_t r = 0;
    #pragma unroll
    for (int e = 0; e < 2; ++e) {
        int cin = kc * 32 + kbase + e;
        __bf16 hb = (__bf16)w[((size_t)m * CCH + cin) * 27 + tap];
        r |= (uint32_t)__builtin_bit_cast(unsigned short, hb) << (16 * e);
    }
    wpack[t] = r;
}

__global__ void k_pack_mlp(const float* __restrict__ w, uint32_t* __restrict__ wpack) {
    int t = blockIdx.x * 256 + threadIdx.x;
    if (t >= 2 * 4 * 32 * 8) return;
    int v    = t & 7;
    int lane = (t >> 3) & 31;
    int mt   = (t >> 8) & 3;
    int kc   = (t >> 10) & 1;
    int m = mt * 16 + (lane & 15);
    int kbase = ((v & 4) << 2) + ((lane >= 16) ? 8 : 0) + 2 * (v & 3);
    uint32_t r = 0;
    #pragma unroll
    for (int e = 0; e < 2; ++e) {
        int cin = kc * 32 + kbase + e;
        __bf16 hb = (__bf16)w[(size_t)m * CCH + cin];
        r |= (uint32_t)__builtin_bit_cast(unsigned short, hb) << (16 * e);
    }
    wpack[t] = r;
}

// ---------------- 3x3x3 conv as implicit GEMM, bf16 WMMA, fused BN+LeakyReLU --------
// grid: B*32*16 WGs; WG tile = 64 Cout x (2 h-rows * 32 w); LDS slab [3][4][34][68] bf16
// wave mapping: 2 cout-pairs x 4 spatial-quarters; each wave: 2 M-tiles x 1 N-tile,
// B fragment loaded once from LDS and reused by two independent WMMAs.
#define CONV_SMEM (3 * 4 * 34 * 68 * 2)
__global__ __launch_bounds__(256) void k_conv3x3_wmma(
    const __bf16* __restrict__ gin,                  // [B][64][32][32][32]
    const uint32_t* __restrict__ wpack,              // packed A fragments
    const float* __restrict__ bias,
    const float* __restrict__ gamma, const float* __restrict__ beta,
    const float* __restrict__ bmean, const float* __restrict__ bvar,
    __bf16* __restrict__ out_bf,                     // mode 0: [B][64][R3]
    float* __restrict__ out_f32,                     // mode 1: [B][R3][64]
    int mode) {
    extern __shared__ __bf16 smem[];                 // [dp3][hr4][wi34][cin 68(pad)]
    const int tid = threadIdx.x;
    const int blk = blockIdx.x;
    const int b  = blk >> 9;
    const int d0 = (blk >> 4) & 31;
    const int h0 = (blk & 15) << 1;

    // ---- stage input slab (zero halo) ----
    for (int j = tid; j < 3 * CCH * 4; j += 256) {
        int dp  = j / (CCH * 4);
        int rem = j - dp * CCH * 4;
        int cin = rem >> 2;
        int hr  = rem & 3;
        int d = d0 + dp - 1;
        int h = h0 + hr - 1;
        __bf16* dst = &smem[((dp * 4 + hr) * 34) * 68 + cin];
        if ((unsigned)d < 32u && (unsigned)h < 32u) {
            const __bf16* src = &gin[(((size_t)(b * CCH + cin) * 32 + d) * 32 + h) * 32];
            dst[0] = (__bf16)0.f;
            Pack8 p;
            #pragma unroll
            for (int q = 0; q < 4; ++q) {
                p.q = ((const uint4*)src)[q];
                #pragma unroll
                for (int e = 0; e < 8; ++e) dst[(q * 8 + e + 1) * 68] = p.h[e];
            }
            dst[33 * 68] = (__bf16)0.f;
        } else {
            for (int wi = 0; wi < 34; ++wi) dst[wi * 68] = (__bf16)0.f;
        }
    }
    __syncthreads();

    const uint32_t* smem_u32 = (const uint32_t*)smem;   // 34 dwords per wi
    const int wave   = tid >> 5;
    const int lane   = tid & 31;
    const int mpair  = wave & 1;                        // couts [mpair*32, +32)
    const int nq     = wave >> 1;                       // spatial [nq*16, +16)
    const int ncol   = lane & 15;
    const int kshift = (lane >= 16) ? 4 : 0;            // dword shift for K+8

    const int n  = nq * 16 + ncol;                      // loop-invariant spatial pos
    const int hr = n >> 5, w = n & 31;
    const int sbase = (hr * 34 + w) * 34 + kshift;      // dword idx, geometry folded below

    floatx8 acc0 = {}, acc1 = {};
    // wpack walk: index (((tap*2+kc)*4 + mt)*32 + lane)*8 ; stride 1024 dwords per (tap,kc)
    const uint32_t* wp0 = wpack + ((size_t)(mpair * 2) * 32 + lane) * 8;
    int tc = 0;
    for (int dd = 0; dd < 3; ++dd) {
        for (int dh = 0; dh < 3; ++dh) {
            const int srow = ((dd * 4 + dh) * 34) * 34 + sbase;
            for (int dw = 0; dw < 3; ++dw) {
                const int sgeo = srow + dw * 34;
                #pragma unroll
                for (int kc = 0; kc < 2; ++kc, ++tc) {
                    const uint32_t* wp = wp0 + (size_t)tc * 1024;
                    __builtin_prefetch(wp + 2 * 1024, 0, 3);   // next (tap,kc) A frags
                    FragAB A0, A1, Bf;
                    A0.q[0] = ((const uint4*)wp)[0];
                    A0.q[1] = ((const uint4*)wp)[1];
                    A1.q[0] = ((const uint4*)(wp + 256))[0];
                    A1.q[1] = ((const uint4*)(wp + 256))[1];
                    const uint32_t* bp = &smem_u32[sgeo + kc * 16];
                    #pragma unroll
                    for (int v = 0; v < 8; ++v)
                        Bf.u[v] = bp[((v & 4) << 1) + (v & 3)];
                    acc0 = __builtin_amdgcn_wmma_f32_16x16x32_bf16(
                        false, A0.v, false, Bf.v, (short)0, acc0, false, false);
                    acc1 = __builtin_amdgcn_wmma_f32_16x16x32_bf16(
                        false, A1.v, false, Bf.v, (short)0, acc1, false, false);
                }
            }
        }
    }

    // ---- epilogue: BN fold + LeakyReLU, store ----
    const int half = lane >> 4;
    const int vox  = d0 * 1024 + (h0 + hr) * 32 + w;
    #pragma unroll
    for (int mt = 0; mt < 2; ++mt) {
        const floatx8& a = mt ? acc1 : acc0;
        #pragma unroll
        for (int r = 0; r < 8; ++r) {
            int cout = (mpair * 2 + mt) * 16 + half * 8 + r;
            float s = gamma[cout] * rsqrtf(bvar[cout] + BN_EPS);
            float t = beta[cout] + (bias[cout] - bmean[cout]) * s;
            float y = a[r] * s + t;
            y = (y >= 0.f) ? y : LEAK * y;
            if (mode == 0)
                out_bf[((size_t)(b * CCH + cout)) * R3 + vox] = (__bf16)y;
            else
                out_f32[((size_t)b * R3 + vox) * CCH + cout] = y;
        }
    }
}

// ---------------- point MLP (64x64) via WMMA, fused BN+ReLU, writes p -> d_out -----
__global__ __launch_bounds__(256) void k_mlp_wmma(
    const float* __restrict__ feat,                  // [B][64][N]
    const uint32_t* __restrict__ wpack,              // [2][4][32][8]
    const float* __restrict__ bias,
    const float* __restrict__ gamma, const float* __restrict__ beta,
    const float* __restrict__ bmean, const float* __restrict__ bvar,
    float* __restrict__ out) {                       // [B][64][N]
    __shared__ __bf16 smem[128 * 68];                // [pt][cin pad 68]
    const int tid = threadIdx.x;
    const int blk = blockIdx.x;                      // B * N/128
    const int b  = blk >> 8;
    const int n0 = (blk & 255) * 128;

    for (int j = tid; j < CCH * 128; j += 256) {
        int cin = j >> 7, i = j & 127;
        smem[i * 68 + cin] = (__bf16)feat[((size_t)b * CCH + cin) * NPTS + n0 + i];
    }
    __syncthreads();

    const uint32_t* smem_u32 = (const uint32_t*)smem;
    const int wave = tid >> 5;
    const int lane = tid & 31;
    const int ncol = lane & 15;
    const int kshift = (lane >= 16) ? 4 : 0;
    const int pbase = wave * 16;

    floatx8 acc[4] = {};
    #pragma unroll
    for (int kc = 0; kc < 2; ++kc) {
        FragAB Bf;
        const uint32_t* bp = &smem_u32[(size_t)(pbase + ncol) * 34 + kc * 16 + kshift];
        #pragma unroll
        for (int v = 0; v < 8; ++v) Bf.u[v] = bp[((v & 4) << 1) + (v & 3)];
        #pragma unroll
        for (int mt = 0; mt < 4; ++mt) {
            FragAB A;
            const uint32_t* wp = &wpack[((kc * 4 + mt) * 32 + lane) * 8];
            A.q[0] = ((const uint4*)wp)[0];
            A.q[1] = ((const uint4*)wp)[1];
            acc[mt] = __builtin_amdgcn_wmma_f32_16x16x32_bf16(
                false, A.v, false, Bf.v, (short)0, acc[mt], false, false);
        }
    }

    const int half = lane >> 4;
    const int n = n0 + pbase + ncol;
    #pragma unroll
    for (int mt = 0; mt < 4; ++mt) {
        #pragma unroll
        for (int r = 0; r < 8; ++r) {
            int cout = mt * 16 + half * 8 + r;
            float s = gamma[cout] * rsqrtf(bvar[cout] + BN_EPS);
            float t = beta[cout] + (bias[cout] - bmean[cout]) * s;
            float y = acc[mt][r] * s + t;
            out[((size_t)(b * CCH + cout)) * NPTS + n] = fmaxf(y, 0.f);
        }
    }
}

// ---------------- trilinear devoxelize from channels-last grid, += into out --------
__global__ void k_devox_add(const float* __restrict__ g2,   // [B][R3][64]
                            const float* __restrict__ nc,   // [B*N][3]
                            float* __restrict__ out) {      // [B][64][N]
    int t = blockIdx.x * 256 + threadIdx.x;
    if (t >= BATCH * NPTS) return;
    int b = t / NPTS, n = t - b * NPTS;
    float x = nc[(size_t)t * 3 + 0];
    float y = nc[(size_t)t * 3 + 1];
    float z = nc[(size_t)t * 3 + 2];
    int lx = (int)floorf(x), ly = (int)floorf(y), lz = (int)floorf(z);
    float fx = x - lx, fy = y - ly, fz = z - lz;

    float4 acc[16];
    #pragma unroll
    for (int i = 0; i < 16; ++i) acc[i] = make_float4(0.f, 0.f, 0.f, 0.f);

    #pragma unroll
    for (int tp = 0; tp < 8; ++tp) {
        int dx = tp >> 2, dy = (tp >> 1) & 1, dz = tp & 1;
        int ix = min(lx + dx, 31), iy = min(ly + dy, 31), iz = min(lz + dz, 31);
        float w = (dx ? fx : 1.f - fx) * (dy ? fy : 1.f - fy) * (dz ? fz : 1.f - fz);
        const float4* gp = (const float4*)&g2[((size_t)b * R3 + ((ix * RES + iy) * RES + iz)) * CCH];
        #pragma unroll
        for (int c4 = 0; c4 < 16; ++c4) {
            float4 g = gp[c4];
            acc[c4].x += w * g.x; acc[c4].y += w * g.y;
            acc[c4].z += w * g.z; acc[c4].w += w * g.w;
        }
    }
    #pragma unroll
    for (int c4 = 0; c4 < 16; ++c4) {
        out[((size_t)(b * CCH + 4 * c4 + 0)) * NPTS + n] += acc[c4].x;
        out[((size_t)(b * CCH + 4 * c4 + 1)) * NPTS + n] += acc[c4].y;
        out[((size_t)(b * CCH + 4 * c4 + 2)) * NPTS + n] += acc[c4].z;
        out[((size_t)(b * CCH + 4 * c4 + 3)) * NPTS + n] += acc[c4].w;
    }
}

// =====================================================================
extern "C" void kernel_launch(void* const* d_in, const int* in_sizes, int n_in,
                              void* d_out, int out_size, void* d_ws, size_t ws_size,
                              hipStream_t stream) {
    const float* features = (const float*)d_in[0];
    const float* coords   = (const float*)d_in[1];
    // d_in[2] = temb (unused by the reference)
    const float* conv1_w = (const float*)d_in[3];
    const float* conv1_b = (const float*)d_in[4];
    const float* bn1_g = (const float*)d_in[5], *bn1_b = (const float*)d_in[6];
    const float* bn1_m = (const float*)d_in[7], *bn1_v = (const float*)d_in[8];
    const float* conv2_w = (const float*)d_in[9];
    const float* conv2_b = (const float*)d_in[10];
    const float* bn2_g = (const float*)d_in[11], *bn2_b = (const float*)d_in[12];
    const float* bn2_m = (const float*)d_in[13], *bn2_v = (const float*)d_in[14];
    const float* mlp_w = (const float*)d_in[15];
    const float* mlp_b = (const float*)d_in[16];
    const float* bnp_g = (const float*)d_in[17], *bnp_b = (const float*)d_in[18];
    const float* bnp_m = (const float*)d_in[19], *bnp_v = (const float*)d_in[20];
    float* out = (float*)d_out;

    // ---- workspace carve-up (256B aligned regions) ----
    char* p = (char*)d_ws;
    auto carve = [&](size_t bytes) -> void* {
        void* r = (void*)p;
        p += (bytes + 255) & ~(size_t)255;
        return r;
    };
    float*    mean   = (float*)carve((size_t)BATCH * 3 * 4);
    float*    rscale = (float*)carve((size_t)BATCH * 4);
    float*    nc     = (float*)carve((size_t)BATCH * NPTS * 3 * 4);
    float*    cnt    = (float*)carve((size_t)BATCH * R3 * 4);            // 1 MB
    float*    gsum   = (float*)carve((size_t)BATCH * R3 * CCH * 4);      // 64 MB, reused as conv2 out
    __bf16*   g0     = (__bf16*)carve((size_t)BATCH * CCH * R3 * 2);     // 32 MB
    __bf16*   g1     = (__bf16*)carve((size_t)BATCH * CCH * R3 * 2);     // 32 MB
    uint32_t* wp1    = (uint32_t*)carve((size_t)55296 * 4);
    uint32_t* wp2    = (uint32_t*)carve((size_t)55296 * 4);
    uint32_t* wpm    = (uint32_t*)carve((size_t)2048 * 4);

    // ---- voxelize ----
    k_coord_mean<<<BATCH, 256, 0, stream>>>(coords, mean);
    k_coord_scale<<<BATCH, 256, 0, stream>>>(coords, mean, rscale);
    // zero cnt + gsum in one contiguous memset (regions are adjacent)
    hipMemsetAsync(cnt, 0, (size_t)BATCH * R3 * 4 + (size_t)BATCH * R3 * CCH * 4, stream);
    k_voxelize<<<(BATCH * NPTS) / 256, 256, 0, stream>>>(features, coords, mean, rscale, nc, gsum, cnt);
    k_finalize_grid<<<(BATCH * R3) / 256, 256, 0, stream>>>(gsum, cnt, g0);

    // ---- pack weights into WMMA A-fragment layout ----
    k_pack_conv<<<(55296 + 255) / 256, 256, 0, stream>>>(conv1_w, wp1);
    k_pack_conv<<<(55296 + 255) / 256, 256, 0, stream>>>(conv2_w, wp2);
    k_pack_mlp<<<(2048 + 255) / 256, 256, 0, stream>>>(mlp_w, wpm);

    // ---- conv1: bf16 in -> bf16 grid [B][C][R3] ----
    k_conv3x3_wmma<<<BATCH * 32 * 16, 256, CONV_SMEM, stream>>>(
        g0, wp1, conv1_b, bn1_g, bn1_b, bn1_m, bn1_v, g1, (float*)nullptr, 0);
    // ---- conv2: bf16 in -> f32 channels-last grid [B][R3][C] (aliases gsum) ----
    k_conv3x3_wmma<<<BATCH * 32 * 16, 256, CONV_SMEM, stream>>>(
        g1, wp2, conv2_b, bn2_g, bn2_b, bn2_m, bn2_v, (__bf16*)nullptr, gsum, 1);

    // ---- point MLP -> writes p into d_out, then trilinear devox adds ----
    k_mlp_wmma<<<(BATCH * NPTS) / 128, 256, 0, stream>>>(
        features, wpm, mlp_b, bnp_g, bnp_b, bnp_m, bnp_v, out);
    k_devox_add<<<(BATCH * NPTS) / 256, 256, 0, stream>>>(gsum, nc, out);
}
